// KPE2ModuleVotes_38044820308012
// MI455X (gfx1250) — compile-verified, matrix operand order checked by reading, not run on hardware
//
#include <hip/hip_runtime.h>
#include <hip/hip_bf16.h>

typedef __attribute__((ext_vector_type(16))) _Float16 v16h;
typedef __attribute__((ext_vector_type(8)))  _Float16 v8h;
typedef __attribute__((ext_vector_type(4)))  _Float16 v4h;
typedef __attribute__((ext_vector_type(8)))  float    v8f;

#define BB 2
#define NN 4096
#define NPP 2048
#define NSS 32
#define NROT 12
#define KKP 15
#define DOUT 128
#define NRM 49152            // B*NROT*NP rows for both GEMMs
#define MTILES 3072          // NRM/16
#define R2C 0.04f            // RADIUS^2
#define KPE 0.08f            // RADIUS/2.5
#define EPSC 1e-5f
#define SLOPEC 0.1f

// output element offsets (floats/ints) in d_out
#define OUT_XYZ   0
#define OUT_FEAT  12288
#define OUT_INDS  6303744
#define OUT_BQ    6307840

__device__ __forceinline__ v8f wmma_f16(v16h a, v16h b, v8f c) {
  return __builtin_amdgcn_wmma_f32_16x16x32_f16(false, a, false, b, (short)0, c, false, false);
}
__device__ __forceinline__ v16h cat8(v8h lo, v8h hi) {
  return __builtin_shufflevector(lo, hi, 0,1,2,3,4,5,6,7,8,9,10,11,12,13,14,15);
}
__device__ __forceinline__ float leakyf(float x) { return x >= 0.f ? x : SLOPEC * x; }

// ---------------------------------------------------------------- K0: prep
// rotation inverses, rotated kernel points, fragment-swizzled B matrices
__global__ void k0_prep(const float* __restrict__ kernels,  // (15,4,128)
                        const float* __restrict__ kp_base,  // (15,3)
                        const float* __restrict__ conv_w,   // (128,128,3)
                        float* __restrict__ rinv, float* __restrict__ kp,
                        _Float16* __restrict__ bf1, _Float16* __restrict__ bf2) {
  int tid = threadIdx.x;
  if (tid < NROT) {
    int r = tid;
    float ang = (float)r / (float)NROT * 6.28318530717958647692f;
    float c = cosf(ang), s = sinf(ang);
    float R[3][3] = {{c, -s, 0.f}, {s, c, 0.f}, {0.f, 0.f, 1.f}};
    // Rinv = R^T
    for (int i = 0; i < 3; ++i)
      for (int j = 0; j < 3; ++j) rinv[r*9 + i*3 + j] = R[j][i];
    for (int k = 0; k < KKP; ++k)
      for (int i = 0; i < 3; ++i) {
        float acc = 0.f;
        for (int j = 0; j < 3; ++j) acc += R[i][j] * kp_base[k*3 + j];
        kp[(r*KKP + k)*3 + i] = acc;
      }
  }
  // bf1: GEMM1 B (64x128 padded from kernels 60x128), layout [ct][s][lane][16]
  for (int e = tid; e < 8*2*32*16; e += blockDim.x) {
    int h = e & 15, lane = (e >> 4) & 31, s = (e >> 9) & 1, ct = e >> 10;
    int col = ct*16 + (lane & 15);
    int base = s*32 + ((lane < 16) ? 0 : 8);
    int kd = (h < 8) ? (base + h) : (base + 16 + (h - 8));
    float v = (kd < 60) ? kernels[(kd >> 2)*512 + (kd & 3)*128 + col] : 0.f;
    bf1[e] = (_Float16)v;
  }
  // bf2: GEMM2 B (384x128 from conv_w), kd = tau*128+fi, layout [ct][ks][lane][16]
  for (int e = tid; e < 8*12*32*16; e += blockDim.x) {
    int h = e & 15, lane = (e >> 4) & 31;
    int t2 = e >> 9;
    int ks = t2 % 12, ct = t2 / 12;
    int base = ks*32 + ((lane < 16) ? 0 : 8);
    int kd = (h < 8) ? (base + h) : (base + 16 + (h - 8));
    int tau = kd >> 7, fi = kd & 127;
    int fo = ct*16 + (lane & 15);
    bf2[e] = (_Float16)conv_w[fo*384 + fi*3 + tau];
  }
}

// ------------------------------------------------------------- K_misc: new_xyz + inds
__global__ void k_misc(const float* __restrict__ xyz, float* __restrict__ out,
                       int* __restrict__ iout) {
  int tid = blockIdx.x * blockDim.x + threadIdx.x;
  if (tid < 12288) {
    int b = tid / 6144, rem = tid % 6144;
    int n = rem / 3, c = rem % 3;
    out[OUT_XYZ + tid] = xyz[(b*NN + n)*3 + c];
  } else if (tid < 16384) {
    int i = tid - 12288;
    iout[OUT_INDS + i] = i & (NPP - 1);
  }
}

// ------------------------------------------------------------- K1: ball query
// one wave per query point; ordered first-32-in-radius via ballot + prefix rank
__global__ void k1_bq(const float* __restrict__ xyz, int* __restrict__ bq,
                      int* __restrict__ bq_out) {
  int lane = threadIdx.x & 31;
  int qid = blockIdx.x * 8 + (threadIdx.x >> 5);
  int b = qid >> 11, n = qid & (NPP - 1);
  float qx = xyz[(b*NN + n)*3 + 0];
  float qy = xyz[(b*NN + n)*3 + 1];
  float qz = xyz[(b*NN + n)*3 + 2];
  int cnt = 0, first = n;
  for (int j0 = 0; j0 < NN; j0 += 32) {
    int j = j0 + lane;
    float dx = xyz[(b*NN + j)*3 + 0] - qx;
    float dy = xyz[(b*NN + j)*3 + 1] - qy;
    float dz = xyz[(b*NN + j)*3 + 2] - qz;
    bool pred = (dx*dx + dy*dy + dz*dz) <= R2C;
    unsigned int mask = __builtin_amdgcn_ballot_w32(pred);
    if (mask) {
      int total = __popc(mask);
      int rank = __popc(mask & ((1u << lane) - 1u));
      int slot = cnt + rank;
      if (pred && slot < NSS) {
        bq[qid*NSS + slot] = j;
        bq_out[OUT_BQ + qid*NSS + slot] = j;
      }
      if (cnt == 0) first = __shfl(j, __ffs(mask) - 1);
      cnt += total;
      if (cnt >= NSS) { cnt = NSS; break; }
    }
  }
  if (lane >= cnt) {   // pad with first neighbor (reference semantics)
    bq[qid*NSS + lane] = first;
    bq_out[OUT_BQ + qid*NSS + lane] = first;
  }
}

// ------------------------------------------------------------- K2a: build t (f16)
// t[(b*12+r)*2048+n][kd], kd = 4k+d (60 real, 4 zero-pad). One wave per (b,n).
__global__ void k2a_t(const float* __restrict__ xyz, const float* __restrict__ feats,
                      const int* __restrict__ bq, const float* __restrict__ rinv,
                      const float* __restrict__ kp, _Float16* __restrict__ t) {
  __shared__ float sG[8][32][3];
  __shared__ float sF[8][32];
  __shared__ float sRC[8][32][3];
  int tid = threadIdx.x, lane = tid & 31, w = tid >> 5;
  int wv = blockIdx.x * 8 + w;
  int b = wv >> 11, n = wv & (NPP - 1);
  int idx = bq[(b*NPP + n)*NSS + lane];
  float qx = xyz[(b*NN + n)*3 + 0], qy = xyz[(b*NN + n)*3 + 1], qz = xyz[(b*NN + n)*3 + 2];
  sG[w][lane][0] = xyz[(b*NN + idx)*3 + 0] - qx;
  sG[w][lane][1] = xyz[(b*NN + idx)*3 + 1] - qy;
  sG[w][lane][2] = xyz[(b*NN + idx)*3 + 2] - qz;
  sF[w][lane] = feats[b*NN + idx];
  __syncthreads();
  int k = lane & 15, mh = lane >> 4;
  for (int r = 0; r < NROT; ++r) {
    const float* Ri = rinv + r*9;
    float gx = sG[w][lane][0], gy = sG[w][lane][1], gz = sG[w][lane][2];
    sRC[w][lane][0] = Ri[0]*gx + Ri[1]*gy + Ri[2]*gz;
    sRC[w][lane][1] = Ri[3]*gx + Ri[4]*gy + Ri[5]*gz;
    sRC[w][lane][2] = Ri[6]*gx + Ri[7]*gy + Ri[8]*gz;
    __syncthreads();
    float a0 = 0.f, a1 = 0.f, a2 = 0.f, a3 = 0.f;
    if (k < KKP) {
      float kx = kp[(r*KKP + k)*3 + 0];
      float ky = kp[(r*KKP + k)*3 + 1];
      float kz = kp[(r*KKP + k)*3 + 2];
      for (int mm = 0; mm < 16; ++mm) {
        int m = mh*16 + mm;
        float dx = kx - sG[w][m][0];
        float dy = ky - sG[w][m][1];
        float dz = kz - sG[w][m][2];
        float d = sqrtf(dx*dx + dy*dy + dz*dz);
        float wg = 1.f - d * (1.f / KPE);
        wg = wg > 0.f ? wg : 0.f;
        a0 += wg * sRC[w][m][0];
        a1 += wg * sRC[w][m][1];
        a2 += wg * sRC[w][m][2];
        a3 += wg * sF[w][m];
      }
    }
    a0 += __shfl_xor(a0, 16); a1 += __shfl_xor(a1, 16);
    a2 += __shfl_xor(a2, 16); a3 += __shfl_xor(a3, 16);
    if (lane < 16) {
      int row = (b*NROT + r)*NPP + n;
      v4h p;
      p.x = (_Float16)a0; p.y = (_Float16)a1; p.z = (_Float16)a2; p.w = (_Float16)a3;
      *(v4h*)(t + row*64 + k*4) = p;   // k==15 wrote zeros -> pad kd 60..63
    }
    __syncthreads();
  }
}

// ------------------------------------------------------------- K2b: GEMM1 (WMMA)
// nf[row][f] = t(49152x64,f16) x bf1(64x128,f16), f32 accumulate
__global__ void k2b_gemm1(const _Float16* __restrict__ t, const _Float16* __restrict__ bf1,
                          float* __restrict__ nf) {
  int lane = threadIdx.x & 31;
  int wv = blockIdx.x * 8 + (threadIdx.x >> 5);
  if (wv >= MTILES) return;
  int row0 = wv * 16;
  const _Float16* tr = t + (row0 + (lane & 15)) * 64;
  int ksel = (lane < 16) ? 0 : 8;
  v16h A0 = cat8(*(const v8h*)(tr + ksel),      *(const v8h*)(tr + 16 + ksel));
  v16h A1 = cat8(*(const v8h*)(tr + 32 + ksel), *(const v8h*)(tr + 48 + ksel));
  int rbase = row0 + ((lane < 16) ? 0 : 8);
  for (int ct = 0; ct < 8; ++ct) {
    v16h B0 = *(const v16h*)(bf1 + ((ct*2 + 0)*32 + lane)*16);
    v16h B1 = *(const v16h*)(bf1 + ((ct*2 + 1)*32 + lane)*16);
    v8f c = {};
    c = wmma_f16(A0, B0, c);
    c = wmma_f16(A1, B1, c);
    int col = ct*16 + (lane & 15);
#pragma unroll
    for (int i = 0; i < 8; ++i) nf[(rbase + i)*DOUT + col] = c[i];
  }
}

// ------------------------------------------------------------- stats: per-column mean/var
__global__ void k_stats(const float* __restrict__ data, float* __restrict__ stats) {
  int f = blockIdx.x, tid = threadIdx.x;
  float s = 0.f, s2 = 0.f;
  for (int row = tid; row < NRM; row += 256) {
    float v = data[row*DOUT + f];
    s += v; s2 += v*v;
  }
  __shared__ float sh[256], sh2[256];
  sh[tid] = s; sh2[tid] = s2;
  __syncthreads();
  for (int o = 128; o > 0; o >>= 1) {
    if (tid < o) { sh[tid] += sh[tid + o]; sh2[tid] += sh2[tid + o]; }
    __syncthreads();
  }
  if (tid == 0) {
    float m = sh[0] / (float)NRM;
    stats[f] = m;
    stats[DOUT + f] = sh2[0] / (float)NRM - m*m;
  }
}

// ------------------------------------------------------------- K4a: BN1 + leaky -> f16, relayout
__global__ void k4a_bn1(const float* __restrict__ nf, const float* __restrict__ stats1,
                        const float* __restrict__ gamma1, const float* __restrict__ beta1,
                        _Float16* __restrict__ yl) {
  int tid = blockIdx.x * blockDim.x + threadIdx.x;
  if (tid >= NRM * DOUT) return;
  int f = tid & 127, row = tid >> 7;
  int n = row & (NPP - 1), t2 = row >> 11;
  int b = t2 / NROT, r = t2 % NROT;
  float x = nf[row*DOUT + f];
  float y = gamma1[f] * (x - stats1[f]) * rsqrtf(stats1[DOUT + f] + EPSC) + beta1[f];
  y = leakyf(y);
  yl[((b*NPP + n)*NROT + r)*DOUT + f] = (_Float16)y;
}

// ------------------------------------------------------------- K4b: GEMM2 (WMMA)
// c[(bn*12+r)][fo] = sum_{tau,fi} y[bn][(r+tau-1)%12][fi] * conv_w[fo][fi][tau] + bias
__global__ void k4b_gemm2(const _Float16* __restrict__ yl, const _Float16* __restrict__ bf2,
                          const float* __restrict__ conv_b, float* __restrict__ cbuf) {
  int lane = threadIdx.x & 31;
  int wv = blockIdx.x * 8 + (threadIdx.x >> 5);
  if (wv >= MTILES) return;
  int row0 = wv * 16;
  int row = row0 + (lane & 15);
  int bn = row / NROT, r = row % NROT;
  v8f acc[8] = {};
  for (int ks = 0; ks < 12; ++ks) {
    int c0 = ks*32 + ((lane < 16) ? 0 : 8);
    int c1 = c0 + 16;
    int tau0 = c0 >> 7, fi0 = c0 & 127;
    int tau1 = c1 >> 7, fi1 = c1 & 127;
    int rr0 = (r + tau0 + 11) % 12;
    int rr1 = (r + tau1 + 11) % 12;
    v8h lo = *(const v8h*)(yl + (bn*NROT + rr0)*DOUT + fi0);
    v8h hi = *(const v8h*)(yl + (bn*NROT + rr1)*DOUT + fi1);
    v16h A = cat8(lo, hi);
#pragma unroll
    for (int ct = 0; ct < 8; ++ct) {
      v16h Bf = *(const v16h*)(bf2 + ((ct*12 + ks)*32 + lane)*16);
      acc[ct] = wmma_f16(A, Bf, acc[ct]);
    }
  }
  int rbase = row0 + ((lane < 16) ? 0 : 8);
#pragma unroll
  for (int ct = 0; ct < 8; ++ct) {
    int col = ct*16 + (lane & 15);
    float bias = conv_b[col];
#pragma unroll
    for (int i = 0; i < 8; ++i) cbuf[(rbase + i)*DOUT + col] = acc[ct][i] + bias;
  }
}

// ------------------------------------------------------------- K6: BN2 + leaky + transpose out
__global__ void k6_out(const float* __restrict__ cbuf, const float* __restrict__ stats2,
                       const float* __restrict__ gamma2, const float* __restrict__ beta2,
                       float* __restrict__ out) {
  int o = blockIdx.x * blockDim.x + threadIdx.x;
  if (o >= BB * DOUT * NROT * NPP) return;
  int n = o & (NPP - 1);
  int q = o >> 11;
  int r = q % NROT;
  int q2 = q / NROT;
  int f = q2 & 127, b = q2 >> 7;
  float x = cbuf[((b*NPP + n)*NROT + r)*DOUT + f];
  float y = gamma2[f] * (x - stats2[f]) * rsqrtf(stats2[DOUT + f] + EPSC) + beta2[f];
  out[OUT_FEAT + o] = leakyf(y);
}

// ================================================================ host launcher
extern "C" void kernel_launch(void* const* d_in, const int* in_sizes, int n_in,
                              void* d_out, int out_size, void* d_ws, size_t ws_size,
                              hipStream_t stream) {
  const float* xyz     = (const float*)d_in[0];
  const float* feats   = (const float*)d_in[1];
  const float* kernels = (const float*)d_in[2];
  const float* kp_base = (const float*)d_in[3];
  const float* gamma1  = (const float*)d_in[4];
  const float* beta1   = (const float*)d_in[5];
  const float* conv_w  = (const float*)d_in[6];
  const float* conv_b  = (const float*)d_in[7];
  const float* gamma2  = (const float*)d_in[8];
  const float* beta2   = (const float*)d_in[9];

  char* ws = (char*)d_ws;
  float*    rinv = (float*)(ws + 0);          //   432 B
  float*    kp   = (float*)(ws + 1024);       //  2160 B
  float*    st1  = (float*)(ws + 4096);       //  1024 B
  float*    st2  = (float*)(ws + 8192);       //  1024 B
  _Float16* bf1  = (_Float16*)(ws + 16384);   //  16 KB
  _Float16* bf2  = (_Float16*)(ws + 32768);   //  96 KB
  int*      bq   = (int*)(ws + 131072);       // 512 KB
  // t (6.3 MB) and cbuf (25.2 MB) share a region: disjoint lifetimes
  _Float16* t    = (_Float16*)(ws + 655360);
  float*    cbuf = (float*)(ws + 655360);     // ends 25,821,184
  float*    nf   = (float*)(ws + 25821184);   // 25.2 MB, ends 50,987,008
  _Float16* yl   = (_Float16*)(ws + 50987008);// 12.6 MB, ends 63,569,920
  if (ws_size < (size_t)63569920) return;

  float* out = (float*)d_out;
  int*   iout = (int*)d_out;

  k0_prep<<<1, 256, 0, stream>>>(kernels, kp_base, conv_w, rinv, kp, bf1, bf2);
  k_misc<<<64, 256, 0, stream>>>(xyz, out, iout);
  k1_bq<<<512, 256, 0, stream>>>(xyz, bq, iout);
  k2a_t<<<512, 256, 0, stream>>>(xyz, feats, bq, rinv, kp, t);
  k2b_gemm1<<<384, 256, 0, stream>>>(t, bf1, nf);
  k_stats<<<128, 256, 0, stream>>>(nf, st1);
  k4a_bn1<<<24576, 256, 0, stream>>>(nf, st1, gamma1, beta1, yl);
  k4b_gemm2<<<384, 256, 0, stream>>>(yl, bf2, conv_b, cbuf);
  k_stats<<<128, 256, 0, stream>>>(cbuf, st2);
  k6_out<<<24576, 256, 0, stream>>>(cbuf, st2, gamma2, beta2, out);
}
